// LlamaDecoderLayer_29592324669776
// MI455X (gfx1250) — matrix operand order repaired and checked by reading, not compiled, not run on hardware
//
#include <hip/hip_runtime.h>
#include <hip/hip_bf16.h>

// ---------------------------------------------------------------------------
// MI455X (gfx1250) Llama decoder layer, compile-only target.
// Roofline: ~1.25 TFLOP vs ~850MB unavoidable HBM traffic @ 23.3 TB/s
// -> memory bound (~35us floor). Strategy: single pass over fp32 weights
// (convert to bf16 while staging into LDS, float2 loads), bf16 activations,
// f32 WMMA accumulate (v_wmma_f32_16x16x32_bf16), fused epilogues
// (residual adds, silu*up), and TDM (tensor_load_to_lds) for the A-tile
// so the activation copy overlaps with weight conversion.
// ---------------------------------------------------------------------------

typedef __attribute__((ext_vector_type(16))) __bf16 v16bf;
typedef __attribute__((ext_vector_type(8)))  __bf16 v8bf;
typedef __attribute__((ext_vector_type(8)))  float  v8f;
typedef __attribute__((ext_vector_type(4)))  unsigned int v4u;
typedef __attribute__((ext_vector_type(4)))  int v4i;
typedef __attribute__((ext_vector_type(8)))  int v8i;

#define BM 64
#define BN 128
#define BK 32

#if defined(__HIP_DEVICE_COMPILE__) && \
    __has_builtin(__builtin_amdgcn_tensor_load_to_lds) && \
    __has_builtin(__builtin_amdgcn_s_wait_tensorcnt)
#define USE_TDM 1
#else
#define USE_TDM 0
#endif

// ---------------------------------------------------------------------------
// RMSNorm: one block per token row; f32 in, bf16 out (A-matrix for WMMA).
// ---------------------------------------------------------------------------
__global__ __launch_bounds__(256) void rmsnorm_bf16_kernel(
    const float* __restrict__ x, const float* __restrict__ w,
    unsigned short* __restrict__ out_raw, int D)
{
    __bf16* out = (__bf16*)out_raw;
    const int row = blockIdx.x;
    const float* xr = x + (size_t)row * D;
    __bf16* outr = out + (size_t)row * D;

    float ss = 0.f;
    const float4* x4 = (const float4*)xr;
    const int n4 = D >> 2;
    for (int i = threadIdx.x; i < n4; i += 256) {
        float4 v = x4[i];
        ss += v.x * v.x + v.y * v.y + v.z * v.z + v.w * v.w;
    }
    __shared__ float red[256];
    red[threadIdx.x] = ss;
    __syncthreads();
    for (int s = 128; s > 0; s >>= 1) {
        if (threadIdx.x < s) red[threadIdx.x] += red[threadIdx.x + s];
        __syncthreads();
    }
    const float scale = rsqrtf(red[0] / (float)D + 1e-6f);

    const float4* w4 = (const float4*)w;
    struct bf16x4 { __bf16 a, b, c, d; };
    bf16x4* o4 = (bf16x4*)outr;
    for (int i = threadIdx.x; i < n4; i += 256) {
        float4 v = x4[i];
        float4 g = w4[i];
        bf16x4 r;
        r.a = (__bf16)(v.x * scale * g.x);
        r.b = (__bf16)(v.y * scale * g.y);
        r.c = (__bf16)(v.z * scale * g.z);
        r.d = (__bf16)(v.w * scale * g.w);
        o4[i] = r;
    }
}

// ---------------------------------------------------------------------------
// TDM: DMA a [tile_m x tile_k] bf16 tile (row-major, row stride = tensor_k
// elements) from global memory into LDS at byte offset lds_off.
// D# layout per CDNA5 ISA:
//   group0: count=1 | lds_addr | global_addr[56:0] | type=2
//   group1: data_size=1(2B) | tensor_dim0/1 | tile_dim0/1 | dim0_stride
// Issued by one wave; completion via TENSORcnt (s_wait_tensorcnt 0).
// This toolchain exposes the 6-arg builtin form (extra descriptor group).
// ---------------------------------------------------------------------------
#if USE_TDM
__device__ inline void tdm_load_2d_bf16(unsigned lds_off, const void* gptr,
                                        unsigned tensor_k, unsigned tensor_m,
                                        unsigned tile_k, unsigned tile_m)
{
    const unsigned long long ga = (unsigned long long)gptr;
    v4u g0;
    g0.x = 1u;                                   // count=1 (valid user D#)
    g0.y = lds_off;                              // lds_addr (bytes)
    g0.z = (unsigned)ga;                         // global_addr[31:0]
    g0.w = (unsigned)((ga >> 32) & 0x01FFFFFFu)  // global_addr[56:32]
         | (2u << 30);                           // type=2 ("image")
    v8i g1;
    g1[0] = (int)(1u << 16);                     // data_size=1 -> 2 bytes
    g1[1] = (int)(tensor_k << 16);               // tensor_dim0[15:0]
    g1[2] = (int)((tensor_k >> 16) | (tensor_m << 16));  // dim0[31:16]|dim1[15:0]
    g1[3] = (int)((tensor_m >> 16) | (tile_k << 16));    // dim1[31:16]|tile_dim0
    g1[4] = (int)tile_m;                         // tile_dim1 (tile_dim2=0)
    g1[5] = (int)tensor_k;                       // tensor_dim0_stride[31:0]
    g1[6] = 0;                                   // stride[47:32]|dim1_stride lo
    g1[7] = 0;                                   // dim1_stride hi
    v4i z4 = {0, 0, 0, 0};
    v8i z8 = {0, 0, 0, 0, 0, 0, 0, 0};
    __builtin_amdgcn_tensor_load_to_lds(g0, g1, z4, z4, z8, 0);
}
#endif

// ---------------------------------------------------------------------------
// Fragment helpers (CDNA5 ISA §7.12.2, 16-bit layouts):
//   A 16x32: lane l -> row M=l%16, half h=l/16; K runs [8h,8h+8)+[16+8h,..)
//   B 32x16: lane l -> col N=l%16, half h=l/16; K runs [8h,8h+8)+[16+8h,..)
// ---------------------------------------------------------------------------
__device__ inline v16bf load_a_frag(const __bf16* rowp, int half)
{
    v8bf lo = *(const v8bf*)(rowp + half * 8);
    v8bf hi = *(const v8bf*)(rowp + 16 + half * 8);
    return __builtin_shufflevector(lo, hi, 0,1,2,3,4,5,6,7,8,9,10,11,12,13,14,15);
}

__device__ inline v16bf load_b_frag(const __bf16* colp, int half)
{
    v8bf lo = *(const v8bf*)(colp + half * 8);
    v8bf hi = *(const v8bf*)(colp + 16 + half * 8);
    return __builtin_shufflevector(lo, hi, 0,1,2,3,4,5,6,7,8,9,10,11,12,13,14,15);
}

// Stage B weights: 32(k) x 128(n) fp32 -> bf16, transposed to [n][k] in LDS.
// float2 loads along n (coalesced across lanes, half the VMEM instructions).
__device__ inline void stage_b_tile(__bf16* Bls, const float* __restrict__ B,
                                    size_t rowStride, int k0, int n0, int tid)
{
    const int n  = (tid & 63) * 2;
    const int kb = (tid >> 6) * 8;
    const float* src = B + (size_t)(k0 + kb) * rowStride + n0 + n;
    #pragma unroll
    for (int j = 0; j < 8; ++j) {
        float2 v = *(const float2*)(src + (size_t)j * rowStride);
        Bls[(n + 0) * BK + kb + j] = (__bf16)v.x;
        Bls[(n + 1) * BK + kb + j] = (__bf16)v.y;
    }
}

// ---------------------------------------------------------------------------
// GEMM: Out[M][N] = A(bf16)[M][K] @ B(f32->bf16)[K][N] + Res[M][N]
// Block tile 64x128, 8 waves (2x4), wave tile 32x32 = 2x2 WMMA tiles.
// Dynamic LDS: A tile at offset 0 (TDM target), B tile after it.
// ---------------------------------------------------------------------------
__global__ __launch_bounds__(256) void gemm_res_kernel(
    const unsigned short* __restrict__ A_raw,  // [M][K] bf16
    const float* __restrict__ B,               // [K][N] f32 row-major
    const float* __restrict__ Res,             // [M][N] f32 residual
    float* __restrict__ Out,                   // [M][N] f32
    int M, int N, int K)
{
    const __bf16* A = (const __bf16*)A_raw;
    extern __shared__ char smem[];
    __bf16* Als = (__bf16*)smem;                     // [BM][BK]  4 KB @ off 0
    __bf16* Bls = (__bf16*)(smem + BM * BK * 2);     // [BN][BK]  8 KB

    const int tid  = threadIdx.x;
    const int wave = tid >> 5;
    const int lane = tid & 31;
    const int m0 = blockIdx.x * BM;
    const int n0 = blockIdx.y * BN;
    const int wm = wave & 1;
    const int wn = wave >> 1;
    const int half = lane >> 4;
    const int lr   = lane & 15;

    v8f acc[2][2] = {};

    for (int k0 = 0; k0 < K; k0 += BK) {
        __syncthreads();
#if USE_TDM
        if (wave == 0)  // one DMA per block; overlaps with B staging below
            tdm_load_2d_bf16(0u, A + (size_t)m0 * K + k0,
                             (unsigned)K, (unsigned)M, BK, BM);
#else
        {
            const int idx = tid * 8;
            const int r = idx >> 5;
            const int c = idx & 31;
            *(v8bf*)(Als + r * BK + c) =
                *(const v8bf*)(A + (size_t)(m0 + r) * K + k0 + c);
        }
#endif
        stage_b_tile(Bls, B, (size_t)N, k0, n0, tid);
#if USE_TDM
        if (wave == 0) __builtin_amdgcn_s_wait_tensorcnt(0);
#endif
        __syncthreads();

        v16bf af[2], bf[2];
        #pragma unroll
        for (int i = 0; i < 2; ++i)
            af[i] = load_a_frag(Als + (wm * 32 + i * 16 + lr) * BK, half);
        #pragma unroll
        for (int i = 0; i < 2; ++i)
            bf[i] = load_b_frag(Bls + (wn * 32 + i * 16 + lr) * BK, half);

        #pragma unroll
        for (int mi = 0; mi < 2; ++mi)
            #pragma unroll
            for (int ni = 0; ni < 2; ++ni)
                acc[mi][ni] = __builtin_amdgcn_wmma_f32_16x16x32_bf16(
                    false, af[mi], false, bf[ni], (short)0, acc[mi][ni],
                    false, false);
    }

    // epilogue: C layout -> VGPR j: M = base + 8*half + j, N = lr
    #pragma unroll
    for (int mi = 0; mi < 2; ++mi) {
        #pragma unroll
        for (int ni = 0; ni < 2; ++ni) {
            const int nc = n0 + wn * 32 + ni * 16 + lr;
            #pragma unroll
            for (int j = 0; j < 8; ++j) {
                const int mr = m0 + wm * 32 + mi * 16 + half * 8 + j;
                const size_t off = (size_t)mr * N + nc;
                Out[off] = acc[mi][ni][j] + Res[off];
            }
        }
    }
}

// ---------------------------------------------------------------------------
// Gate/Up GEMM with fused silu(gate)*up epilogue -> bf16 activation.
// Two B tiles (gate cols n0.., up cols n0+I..) per A-tile pass, so the
// 361MB gate_up f32 tensor never round-trips HBM.
// ---------------------------------------------------------------------------
__global__ __launch_bounds__(256) void gemm_gateup_kernel(
    const unsigned short* __restrict__ A_raw,  // [M][K] bf16 (K = D)
    const float* __restrict__ W,               // [K][2I] f32 row-major
    unsigned short* __restrict__ Act_raw,      // [M][I] bf16 out
    int M, int I, int K)
{
    const __bf16* A = (const __bf16*)A_raw;
    __bf16* Act = (__bf16*)Act_raw;
    const size_t N2 = 2 * (size_t)I;

    extern __shared__ char smem[];
    __bf16* Als = (__bf16*)smem;                         // 4 KB @ off 0 (TDM)
    __bf16* Bgl = (__bf16*)(smem + BM * BK * 2);         // 8 KB
    __bf16* Bul = (__bf16*)(smem + BM * BK * 2 + BN * BK * 2); // 8 KB

    const int tid  = threadIdx.x;
    const int wave = tid >> 5;
    const int lane = tid & 31;
    const int m0 = blockIdx.x * BM;
    const int n0 = blockIdx.y * BN;   // within [0, I)
    const int wm = wave & 1;
    const int wn = wave >> 1;
    const int half = lane >> 4;
    const int lr   = lane & 15;

    v8f accg[2][2] = {};
    v8f accu[2][2] = {};

    for (int k0 = 0; k0 < K; k0 += BK) {
        __syncthreads();
#if USE_TDM
        if (wave == 0)
            tdm_load_2d_bf16(0u, A + (size_t)m0 * K + k0,
                             (unsigned)K, (unsigned)M, BK, BM);
#else
        {
            const int idx = tid * 8;
            const int r = idx >> 5;
            const int c = idx & 31;
            *(v8bf*)(Als + r * BK + c) =
                *(const v8bf*)(A + (size_t)(m0 + r) * K + k0 + c);
        }
#endif
        stage_b_tile(Bgl, W, N2, k0, n0, tid);          // gate block
        stage_b_tile(Bul, W + I, N2, k0, n0, tid);      // up block
#if USE_TDM
        if (wave == 0) __builtin_amdgcn_s_wait_tensorcnt(0);
#endif
        __syncthreads();

        v16bf af[2], bg[2], bu[2];
        #pragma unroll
        for (int i = 0; i < 2; ++i)
            af[i] = load_a_frag(Als + (wm * 32 + i * 16 + lr) * BK, half);
        #pragma unroll
        for (int i = 0; i < 2; ++i) {
            bg[i] = load_b_frag(Bgl + (wn * 32 + i * 16 + lr) * BK, half);
            bu[i] = load_b_frag(Bul + (wn * 32 + i * 16 + lr) * BK, half);
        }

        #pragma unroll
        for (int mi = 0; mi < 2; ++mi)
            #pragma unroll
            for (int ni = 0; ni < 2; ++ni) {
                accg[mi][ni] = __builtin_amdgcn_wmma_f32_16x16x32_bf16(
                    false, af[mi], false, bg[ni], (short)0, accg[mi][ni],
                    false, false);
                accu[mi][ni] = __builtin_amdgcn_wmma_f32_16x16x32_bf16(
                    false, af[mi], false, bu[ni], (short)0, accu[mi][ni],
                    false, false);
            }
    }

    // fused silu(gate) * up -> bf16 store
    #pragma unroll
    for (int mi = 0; mi < 2; ++mi) {
        #pragma unroll
        for (int ni = 0; ni < 2; ++ni) {
            const int nc = n0 + wn * 32 + ni * 16 + lr;
            #pragma unroll
            for (int j = 0; j < 8; ++j) {
                const int mr = m0 + wm * 32 + mi * 16 + half * 8 + j;
                const float g = accg[mi][ni][j];
                const float u = accu[mi][ni][j];
                const float s = g / (1.0f + __expf(-g));  // silu
                Act[(size_t)mr * I + nc] = (__bf16)(s * u);
            }
        }
    }
}

// ---------------------------------------------------------------------------
// Host-side launcher
// ---------------------------------------------------------------------------
extern "C" void kernel_launch(void* const* d_in, const int* in_sizes, int n_in,
                              void* d_out, int out_size, void* d_ws, size_t ws_size,
                              hipStream_t stream)
{
    const float* hidden      = (const float*)d_in[0];  // [2,2048,4096]
    const float* in_norm_w   = (const float*)d_in[1];  // [4096]
    const float* o_w         = (const float*)d_in[2];  // [4096,4096]
    const float* post_norm_w = (const float*)d_in[3];  // [4096]
    const float* gate_up_w   = (const float*)d_in[4];  // [4096,22016]
    const float* down_w      = (const float*)d_in[5];  // [11008,4096]
    float* out = (float*)d_out;                        // [2,2048,4096]

    const int Mtok = 2 * 2048;   // 4096 tokens
    const int D    = 4096;
    const int I    = 11008;

    // Workspace layout (~186 MB):
    //   [0, 32MB)      normed activations, bf16 (reused for both norms)
    //   [32MB, 96MB)   x = hidden + attn_out, f32
    //   [96MB, 186MB)  act = silu(gate)*up, bf16
    char* ws = (char*)d_ws;
    unsigned short* a_bf = (unsigned short*)ws;
    float* x             = (float*)(ws + (size_t)Mtok * D * 2);
    unsigned short* act  = (unsigned short*)(ws + (size_t)Mtok * D * 2
                                                + (size_t)Mtok * D * 4);

    const size_t lds_gemm   = (size_t)(BM * BK + BN * BK) * 2;       // 12 KB
    const size_t lds_gateup = (size_t)(BM * BK + 2 * BN * BK) * 2;   // 20 KB

    // 1) h = rmsnorm(hidden, in_norm_w) -> bf16
    rmsnorm_bf16_kernel<<<Mtok, 256, 0, stream>>>(hidden, in_norm_w, a_bf, D);

    // 2) x = hidden + h @ o_w            (M=4096, N=4096, K=4096)
    gemm_res_kernel<<<dim3(Mtok / BM, D / BN), 256, lds_gemm, stream>>>(
        a_bf, o_w, hidden, x, Mtok, D, D);

    // 3) h2 = rmsnorm(x, post_norm_w) -> bf16 (reuse a_bf)
    rmsnorm_bf16_kernel<<<Mtok, 256, 0, stream>>>(x, post_norm_w, a_bf, D);

    // 4) act = silu(h2 @ Wg) * (h2 @ Wu) (M=4096, N=11008, K=4096)
    gemm_gateup_kernel<<<dim3(Mtok / BM, I / BN), 256, lds_gateup, stream>>>(
        a_bf, gate_up_w, act, Mtok, I, D);

    // 5) out = x + act @ down_w          (M=4096, N=4096, K=11008)
    gemm_res_kernel<<<dim3(Mtok / BM, D / BN), 256, lds_gemm, stream>>>(
        act, down_w, x, out, Mtok, D, I);
}
